// EuclideanCodebook_72911364816984
// MI455X (gfx1250) — matrix operand order repaired and checked by reading, not compiled, not run on hardware
//
#include <hip/hip_runtime.h>

#define DECAY 0.8f
#define ONE_MINUS_DECAY 0.2f
#define EPSV 1e-5f

typedef __attribute__((ext_vector_type(16))) __bf16 v16bf;
typedef __attribute__((ext_vector_type(8)))  float  v8f;

__device__ __forceinline__ __bf16 f2bf(float f) { return (__bf16)f; }

// ---------------------------------------------------------------------------
// 0.5 * ||embed_k||^2 per code (pre-scaled for the argmin epilogue)
// ---------------------------------------------------------------------------
__global__ void vq_e2_kernel(const float* __restrict__ embed,
                             float* __restrict__ e2h, int D) {
  int k = blockIdx.x;
  float acc = 0.f;
  for (int d = threadIdx.x; d < D; d += blockDim.x) {
    float v = embed[(size_t)k * D + d];
    acc += v * v;
  }
  __shared__ float s[256];
  s[threadIdx.x] = acc;
  __syncthreads();
  for (int off = blockDim.x >> 1; off > 0; off >>= 1) {
    if ((int)threadIdx.x < off) s[threadIdx.x] += s[threadIdx.x + off];
    __syncthreads();
  }
  if (threadIdx.x == 0) e2h[k] = 0.5f * s[0];
}

// ---------------------------------------------------------------------------
// One-time codebook conversion: f32 -> bf16 hi/lo packed directly in the
// WMMA B-fragment layout. One thread produces one 32-byte fragment pair slot:
//   fragment id = ((tile*8 + chunk)*32 + lane), element i -> K = c*32+(i>>3)*16+half*8+(i&7)
// This removes ALL conversion VALU from the hot distance loop.
// ---------------------------------------------------------------------------
__global__ void vq_pack_embed_kernel(const float* __restrict__ embed,
                                     v16bf* __restrict__ bh_pack,
                                     v16bf* __restrict__ bl_pack,
                                     int K, int D) {
  int t = blockIdx.x * blockDim.x + threadIdx.x;
  int total = (K >> 4) * 8 * 32;
  if (t >= total) return;
  int lane = t & 31;
  int c    = (t >> 5) & 7;
  int tile = t >> 8;
  int half = lane >> 4;
  const float* er = embed + (size_t)(tile * 16 + (lane & 15)) * D;

  v16bf h, l;
#pragma unroll
  for (int i = 0; i < 16; ++i) {
    float v = er[c * 32 + ((i >> 3) << 4) + half * 8 + (i & 7)];
    __bf16 hv = f2bf(v);
    h[i] = hv;
    l[i] = f2bf(v - (float)hv);
  }
  bh_pack[t] = h;
  bl_pack[t] = l;
}

// ---------------------------------------------------------------------------
// Zero accumulators (ws is poisoned; must re-zero every call)
// ---------------------------------------------------------------------------
__global__ void vq_zero_kernel(float* __restrict__ p, size_t n) {
  size_t i = (size_t)blockIdx.x * blockDim.x + threadIdx.x;
  if (i < n) p[i] = 0.f;
}

// ---------------------------------------------------------------------------
// Distance GEMM + argmin via split-bf16 WMMA.
// One wave32 per 16-row x tile; A fragments (hi+lo) resident in VGPRs
// (converted once per row), B fragments streamed pre-packed from L2.
// argmin(d) == argmin(0.5*||e||^2 - x.e)
// ---------------------------------------------------------------------------
__global__ __launch_bounds__(256)
void vq_dist_argmin_kernel(const float* __restrict__ x,
                           const v16bf* __restrict__ bh_pack,
                           const v16bf* __restrict__ bl_pack,
                           const float* __restrict__ e2h,
                           int* __restrict__ ind,
                           int N, int K) {
  const int D = 256;
  const int lane = threadIdx.x & 31;
  const int wave = (int)((blockIdx.x * blockDim.x + threadIdx.x) >> 5);
  const int m0 = wave * 16;
  if (m0 >= N) return;
  const int half = lane >> 4;
  const int l15  = lane & 15;

  // ---- Preload A tile (16 rows x 256) as bf16 hi/lo fragments: 128 VGPRs ----
  // 16-bit A layout: element i -> K = c*32 + (i>>3)*16 + half*8 + (i&7)
  const float* xr = x + (size_t)(m0 + l15) * D;
  v16bf a_hi[8], a_lo[8];
#pragma unroll
  for (int c = 0; c < 8; ++c) {
#pragma unroll
    for (int i = 0; i < 16; ++i) {
      int kk = c * 32 + ((i >> 3) << 4) + half * 8 + (i & 7);
      float v = xr[kk];
      __bf16 h = f2bf(v);
      a_hi[c][i] = h;
      a_lo[c][i] = f2bf(v - (float)h);
    }
  }

  float best[8];
  int   bidx[8];
#pragma unroll
  for (int r = 0; r < 8; ++r) { best[r] = 3.4e38f; bidx[r] = 0; }

  // ---- Stream pre-packed codebook fragments: K/16 tiles of 16 codes ----
  const int ntiles = K >> 4;
  for (int tile = 0; tile < ntiles; ++tile) {
    const v16bf* bh = bh_pack + ((size_t)tile * 8) * 32 + lane;
    const v16bf* bl = bl_pack + ((size_t)tile * 8) * 32 + lane;
    v8f acc = {0.f, 0.f, 0.f, 0.f, 0.f, 0.f, 0.f, 0.f};
#pragma unroll
    for (int c = 0; c < 8; ++c) {
      v16bf bhf = bh[(size_t)c * 32];
      v16bf blf = bl[(size_t)c * 32];
      // x.e ~= xh.eh + xh.el + xl.eh  (error ~2^-16, near-f32 argmin fidelity)
      acc = __builtin_amdgcn_wmma_f32_16x16x32_bf16(false, a_hi[c], false, bhf,
                                                    (short)0, acc, false, false);
      acc = __builtin_amdgcn_wmma_f32_16x16x32_bf16(false, a_hi[c], false, blf,
                                                    (short)0, acc, false, false);
      acc = __builtin_amdgcn_wmma_f32_16x16x32_bf16(false, a_lo[c], false, bhf,
                                                    (short)0, acc, false, false);
    }
    // C/D layout: VGPR r holds (row = r + half*8, col = lane&15)
    const int   col     = (tile << 4) + l15;
    const float half_e2 = e2h[col];
#pragma unroll
    for (int r = 0; r < 8; ++r) {
      float s = half_e2 - acc[r];
      if (s < best[r]) { best[r] = s; bidx[r] = col; }
    }
  }

  // ---- Per-row argmin reduction across the 16 lanes of each half ----
#pragma unroll
  for (int m = 1; m < 16; m <<= 1) {
#pragma unroll
    for (int r = 0; r < 8; ++r) {
      float ov = __shfl_xor(best[r], m, 32);
      int   oi = __shfl_xor(bidx[r], m, 32);
      if (ov < best[r] || (ov == best[r] && oi < bidx[r])) {
        best[r] = ov; bidx[r] = oi;
      }
    }
  }
  if (l15 == 0) {
#pragma unroll
    for (int r = 0; r < 8; ++r) ind[m0 + half * 8 + r] = bidx[r];
  }
}

// ---------------------------------------------------------------------------
// Gather quantize + segment sums (counts, embed_sum) + float indices out
// ---------------------------------------------------------------------------
__global__ void vq_scatter_kernel(const float* __restrict__ x,
                                  const float* __restrict__ embed,
                                  const int* __restrict__ ind,
                                  float* __restrict__ quant,
                                  float* __restrict__ indf,
                                  float* __restrict__ counts,
                                  float* __restrict__ esum, int D) {
  int n = blockIdx.x;
  int k = ind[n];
  for (int d = threadIdx.x; d < D; d += blockDim.x) {
    quant[(size_t)n * D + d] = embed[(size_t)k * D + d];
    atomicAdd(&esum[(size_t)k * D + d], x[(size_t)n * D + d]);
  }
  if (threadIdx.x == 0) {
    atomicAdd(&counts[k], 1.0f);
    indf[n] = (float)k;
  }
}

// ---------------------------------------------------------------------------
// EMA update: cluster_size_new, embed_avg_new, embed_new
// ---------------------------------------------------------------------------
__global__ void vq_update_kernel(const float* __restrict__ cluster_size,
                                 const float* __restrict__ embed_avg,
                                 const float* __restrict__ counts,
                                 const float* __restrict__ esum,
                                 float* __restrict__ out_cs,
                                 float* __restrict__ out_ea,
                                 float* __restrict__ out_en, int D) {
  int k = blockIdx.x;
  float cs = cluster_size[k] * DECAY + ONE_MINUS_DECAY * counts[k];
  if (threadIdx.x == 0) out_cs[k] = cs;
  float inv = 1.0f / (cs + EPSV);
  for (int d = threadIdx.x; d < D; d += blockDim.x) {
    float ea = embed_avg[(size_t)k * D + d] * DECAY +
               ONE_MINUS_DECAY * esum[(size_t)k * D + d];
    out_ea[(size_t)k * D + d] = ea;
    out_en[(size_t)k * D + d] = ea * inv;
  }
}

extern "C" void kernel_launch(void* const* d_in, const int* in_sizes, int n_in,
                              void* d_out, int out_size, void* d_ws, size_t ws_size,
                              hipStream_t stream) {
  const float* x            = (const float*)d_in[0];  // [N, D] flat
  const float* embed        = (const float*)d_in[1];  // [K, D]
  const float* cluster_size = (const float*)d_in[2];  // [K]
  const float* embed_avg    = (const float*)d_in[3];  // [K, D]

  const int K = in_sizes[2];
  const int D = in_sizes[1] / K;      // 256
  const int N = in_sizes[0] / D;      // 65536

  // Workspace layout (all 32B-aligned given the sizes involved)
  float* e2h    = (float*)d_ws;                  // K
  float* counts = e2h + K;                       // K
  float* esum   = counts + K;                    // K*D
  int*   ind    = (int*)(esum + (size_t)K * D);  // N
  v16bf* bh_pack = (v16bf*)(ind + N);            // (K/16)*8*32 fragments
  v16bf* bl_pack = bh_pack + (size_t)(K >> 4) * 8 * 32;

  // Output layout (tuple concatenated flat, all float)
  float* out_q  = (float*)d_out;                // N*D  quantize
  float* out_i  = out_q + (size_t)N * D;        // N    embed_ind (as float)
  float* out_cs = out_i + N;                    // K    cluster_size_new
  float* out_ea = out_cs + K;                   // K*D  embed_avg_new
  float* out_en = out_ea + (size_t)K * D;       // K*D  embed_new

  vq_e2_kernel<<<K, 256, 0, stream>>>(embed, e2h, D);

  int pack_threads = (K >> 4) * 8 * 32;
  vq_pack_embed_kernel<<<(pack_threads + 255) / 256, 256, 0, stream>>>(
      embed, bh_pack, bl_pack, K, D);

  size_t zn = (size_t)K + (size_t)K * D;
  vq_zero_kernel<<<(unsigned)((zn + 255) / 256), 256, 0, stream>>>(counts, zn);

  int waves = N / 16;                 // one wave per 16-row tile
  vq_dist_argmin_kernel<<<(waves + 7) / 8, 256, 0, stream>>>(
      x, bh_pack, bl_pack, e2h, ind, N, K);

  vq_scatter_kernel<<<N, 256, 0, stream>>>(x, embed, ind, out_q, out_i, counts, esum, D);

  vq_update_kernel<<<K, 256, 0, stream>>>(cluster_size, embed_avg, counts, esum,
                                          out_cs, out_ea, out_en, D);
}
